// SwinTransformerClassifier_71047349010629
// MI455X (gfx1250) — matrix operand order repaired and checked by reading, not compiled, not run on hardware
//
#include <hip/hip_runtime.h>
#include <math.h>

// ---------------------------------------------------------------------------
// Swin-B forward for MI455X (gfx1250).  All large GEMMs run through
// v_wmma_f32_16x16x32_bf16 (wave32).  Block tile 64x128, wave tile 32x32
// (2x2 WMMA tiles -> 4 v_wmma per K-step per wave), float4 global loads,
// bf16 staging in LDS.  Bounds handling is hoisted into uniform scalar
// branches so the hot loop is exec-mask free.  ~250 GFLOP @ AI ~300 FLOP/B
// -> compute-bound on the WMMA pipes; elementwise kernels are HBM trivia.
// ---------------------------------------------------------------------------

typedef __attribute__((ext_vector_type(16))) __bf16 v16bf;
typedef __attribute__((ext_vector_type(8)))  __bf16 v8bf;
typedef __attribute__((ext_vector_type(4)))  __bf16 v4bf;
typedef __attribute__((ext_vector_type(8)))  float  v8f;
typedef __attribute__((ext_vector_type(4)))  float  v4f;

#define WIN 7
#define GEMM_GELU 1

__device__ __forceinline__ __bf16 f2bf(float f) {
  unsigned u = __builtin_bit_cast(unsigned, f);
  u += 0x7FFFu + ((u >> 16) & 1u);           // round-to-nearest-even
  unsigned short h = (unsigned short)(u >> 16);
  return __builtin_bit_cast(__bf16, h);
}

__device__ __forceinline__ v4bf pack4(v4f v) {
  v4bf p = { f2bf(v.x), f2bf(v.y), f2bf(v.z), f2bf(v.w) };
  return p;
}

__device__ __forceinline__ float gelu_exact(float x) {
  return 0.5f * x * (1.0f + erff(x * 0.70710678118654752f));
}

// ---------------------------------------------------------------------------
// WMMA GEMM: out[M,N] = act(A[M,K] @ W[N,K]^T + bias) (+ res)
// Grid: x = N/128, y = ceil(M/64).  Block: 256 threads = 8 waves (2x4 grid),
// each wave owns a 32x32 output tile = 2x2 WMMA accumulators.
// Call-site invariants: K % 16 == 0, N % 128 == 0, M % 16 == 0, rows
// 16B-aligned.  M/K tails are handled on a uniform slow path only.
// ---------------------------------------------------------------------------
__global__ void __launch_bounds__(256) wmma_gemm_kernel(
    const float* __restrict__ A, const float* __restrict__ W,
    const float* __restrict__ bias, const float* res,
    float* out, int M, int N, int K, int flags)
{
  __shared__ alignas(64) __bf16 As[64][32];    // [m][k]
  __shared__ alignas(64) __bf16 Bs[128][32];   // [n][k] (W^T columns, K-contig)

  const int tid  = threadIdx.x;
  const int lane = tid & 31;
  const int wave = tid >> 5;
  const int wr   = wave >> 2;        // 0..1 : wave row   (32 rows each)
  const int wc   = wave & 3;         // 0..3 : wave col   (32 cols each)
  const int hi   = lane >> 4;        // lane half
  const int l16  = lane & 15;
  const int m0   = blockIdx.y * 64;
  const int n0   = blockIdx.x * 128;

  // uniform (per-block) fullness of the M / N tile
  const bool mfull = (m0 + 64 <= M);
  const bool nfull = (n0 + 128 <= N);

  // staging coordinates for this thread (row, k-chunk within 32)
  const int arow = tid >> 3;          // +0, +32       (64 rows)
  const int kc   = (tid & 7) * 4;     // 0..28 step 4
  const int brow = tid >> 3;          // +0,+32,+64,+96 (128 rows)

  v8f acc00 = {}, acc01 = {}, acc10 = {}, acc11 = {};

  for (int k0 = 0; k0 < K; k0 += 32) {
    const bool kfull = (k0 + 32 <= K);   // uniform per k-step
    const int  k     = k0 + kc;

    // ---- stage A chunk: 64x32 = 512 float4 chunks, 2 per thread ----------
    if (mfull && kfull) {
#pragma unroll
      for (int j = 0; j < 2; ++j) {
        int row = arow + j * 32;
        v4f v = *(const v4f*)(A + (size_t)(m0 + row) * K + k);
        *(v4bf*)&As[row][kc] = pack4(v);
      }
    } else {
#pragma unroll
      for (int j = 0; j < 2; ++j) {
        int row = arow + j * 32;
        int m = m0 + row;
        v4f v = {};
        if (m < M && k + 4 <= K) v = *(const v4f*)(A + (size_t)m * K + k);
        *(v4bf*)&As[row][kc] = pack4(v);
      }
    }
    // ---- stage B chunk: 128x32 = 1024 float4 chunks, 4 per thread --------
    if (nfull && kfull) {
#pragma unroll
      for (int j = 0; j < 4; ++j) {
        int row = brow + j * 32;
        v4f v = *(const v4f*)(W + (size_t)(n0 + row) * K + k);
        *(v4bf*)&Bs[row][kc] = pack4(v);
      }
    } else {
#pragma unroll
      for (int j = 0; j < 4; ++j) {
        int row = brow + j * 32;
        int n = n0 + row;
        v4f v = {};
        if (n < N && k + 4 <= K) v = *(const v4f*)(W + (size_t)n * K + k);
        *(v4bf*)&Bs[row][kc] = pack4(v);
      }
    }
    __syncthreads();

    // ---- fragments ----
    // A frag (tile t): lane -> row M = wr*32 + t*16 + l16,
    //   halves: K = hi*16 + [0..7]  and  hi*16 + 8 + [0..7]
    v16bf a[2];
#pragma unroll
    for (int t = 0; t < 2; ++t) {
      const int mrow = wr * 32 + t * 16 + l16;
      v8bf alo = *(const v8bf*)&As[mrow][hi * 16];
      v8bf ahi = *(const v8bf*)&As[mrow][hi * 16 + 8];
#pragma unroll
      for (int i = 0; i < 8; ++i) { a[t][i] = alo[i]; a[t][i + 8] = ahi[i]; }
    }
    // B frag (tile u): lane -> col N = wc*32 + u*16 + l16, K = hi*16 + [0..15]
    v16bf b0 = *(const v16bf*)&Bs[wc * 32 + l16][hi * 16];
    v16bf b1 = *(const v16bf*)&Bs[wc * 32 + 16 + l16][hi * 16];

    acc00 = __builtin_amdgcn_wmma_f32_16x16x32_bf16(false, a[0], false, b0, (short)0, acc00, false, false);
    acc01 = __builtin_amdgcn_wmma_f32_16x16x32_bf16(false, a[0], false, b1, (short)0, acc01, false, false);
    acc10 = __builtin_amdgcn_wmma_f32_16x16x32_bf16(false, a[1], false, b0, (short)0, acc10, false, false);
    acc11 = __builtin_amdgcn_wmma_f32_16x16x32_bf16(false, a[1], false, b1, (short)0, acc11, false, false);
    __syncthreads();
  }

  // ---- epilogue: C/D layout: element e -> M = (lane/16)*8 + e, N = lane%16
  v8f* accs[2][2] = { { &acc00, &acc01 }, { &acc10, &acc11 } };
#pragma unroll
  for (int t = 0; t < 2; ++t) {
#pragma unroll
    for (int u = 0; u < 2; ++u) {
      const int n = n0 + wc * 32 + u * 16 + l16;
      const float bv = (bias && n < N) ? bias[n] : 0.0f;
      v8f& av = *accs[t][u];
#pragma unroll
      for (int e = 0; e < 8; ++e) {
        int m = m0 + wr * 32 + t * 16 + hi * 8 + e;
        if ((mfull || m < M) && (nfull || n < N)) {
          float v = av[e] + bv;
          if (flags & GEMM_GELU) v = gelu_exact(v);
          if (res)  v += res[(size_t)m * N + n];
          out[(size_t)m * N + n] = v;
        }
      }
    }
  }
}

// ---------------------------------------------------------------------------
// LayerNorm over last dim; one block per row.
// ---------------------------------------------------------------------------
__global__ void __launch_bounds__(256) layernorm_kernel(
    const float* __restrict__ x, const float* __restrict__ g,
    const float* __restrict__ b, float* __restrict__ out, int C)
{
  __shared__ float rs[256], rs2[256];
  const size_t row = blockIdx.x;
  const float* xr = x + row * C;
  float* orow = out + row * C;
  float s = 0.f, s2 = 0.f;
  for (int c = threadIdx.x; c < C; c += 256) { float v = xr[c]; s += v; s2 += v * v; }
  rs[threadIdx.x] = s; rs2[threadIdx.x] = s2;
  __syncthreads();
  for (int off = 128; off > 0; off >>= 1) {
    if (threadIdx.x < off) {
      rs[threadIdx.x]  += rs[threadIdx.x + off];
      rs2[threadIdx.x] += rs2[threadIdx.x + off];
    }
    __syncthreads();
  }
  float mu  = rs[0] / C;
  float var = rs2[0] / C - mu * mu;
  float inv = rsqrtf(var + 1e-5f);
  for (int c = threadIdx.x; c < C; c += 256)
    orow[c] = (xr[c] - mu) * inv * g[c] + b[c];
}

// ---------------------------------------------------------------------------
// Patch-merge gather (2x2 -> 4C) + LayerNorm, one block per output token.
// Concat order: (0,0),(1,0),(0,1),(1,1).
// ---------------------------------------------------------------------------
__global__ void __launch_bounds__(256) merge_ln_kernel(
    const float* __restrict__ h, const float* __restrict__ g,
    const float* __restrict__ b, float* __restrict__ out, int H, int Wd, int C)
{
  __shared__ float vals[2048];
  __shared__ float rs[256], rs2[256];
  const int W2 = Wd / 2, H2 = H / 2;
  const int idx = blockIdx.x;
  const int j  = idx % W2;
  const int i  = (idx / W2) % H2;
  const int bb = idx / (W2 * H2);
  const int C4 = 4 * C;
  float s = 0.f, s2 = 0.f;
  for (int f = threadIdx.x; f < C4; f += 256) {
    int q = f / C, c = f % C;
    int dy = q & 1, dx = q >> 1;
    float v = h[((size_t)((bb * H + 2 * i + dy) * Wd + 2 * j + dx)) * C + c];
    vals[f] = v; s += v; s2 += v * v;
  }
  rs[threadIdx.x] = s; rs2[threadIdx.x] = s2;
  __syncthreads();
  for (int off = 128; off > 0; off >>= 1) {
    if (threadIdx.x < off) {
      rs[threadIdx.x]  += rs[threadIdx.x + off];
      rs2[threadIdx.x] += rs2[threadIdx.x + off];
    }
    __syncthreads();
  }
  float mu  = rs[0] / C4;
  float var = rs2[0] / C4 - mu * mu;
  float inv = rsqrtf(var + 1e-5f);
  for (int f = threadIdx.x; f < C4; f += 256)
    out[(size_t)idx * C4 + f] = (vals[f] - mu) * inv * g[f] + b[f];
}

// ---------------------------------------------------------------------------
// Window attention core.  One block per (batch*window, head), head_dim = 32
// at every stage.  Shift (roll) folded into gather/scatter indices; mask and
// relative-position bias computed on the fly.
// ---------------------------------------------------------------------------
__global__ void __launch_bounds__(64) window_attn_kernel(
    const float* __restrict__ qkv, const float* __restrict__ bias_table,
    float* __restrict__ attn_out, int H, int Wd, int C, int heads, int shift)
{
  __shared__ float qs[49 * 32], ks[49 * 32], vs[49 * 32];
  const int nW   = Wd / WIN;
  const int nWin = (H / WIN) * nW;
  const int bwi  = blockIdx.x;
  const int b    = bwi / nWin;
  const int wi   = bwi % nWin;
  const int wy   = wi / nW, wx = wi % nW;
  const int head = blockIdx.y;
  const int C3   = 3 * C;

  for (int i = threadIdx.x; i < 49 * 32; i += 64) {
    int tok = i >> 5, d = i & 31;
    int iy = tok / WIN, ix = tok % WIN;
    int ro = (wy * WIN + iy + shift) % H;        // rolled -> original coords
    int co = (wx * WIN + ix + shift) % Wd;
    size_t base = (size_t)((b * H + ro) * Wd + co) * C3 + head * 32 + d;
    qs[i] = qkv[base];
    ks[i] = qkv[base + C];
    vs[i] = qkv[base + 2 * C];
  }
  __syncthreads();

  const int i = threadIdx.x;
  if (i >= 49) return;
  const int yi = i / WIN, xi = i % WIN;

  int li = 0;
  if (shift > 0) {
    int r = wy * WIN + yi, c = wx * WIN + xi;
    int lr = (r < H - WIN) ? 0 : ((r < H - shift) ? 1 : 2);
    int lc = (c < Wd - WIN) ? 0 : ((c < Wd - shift) ? 1 : 2);
    li = lr * 3 + lc;
  }

  float s[49];
  float mx = -1e30f;
  for (int j = 0; j < 49; ++j) {
    float dot = 0.0f;
    for (int d = 0; d < 32; ++d) dot += qs[i * 32 + d] * ks[j * 32 + d];
    dot *= 0.17677669529663687f;                 // 1/sqrt(32)
    int yj = j / WIN, xj = j % WIN;
    int rel = (yi - yj + WIN - 1) * (2 * WIN - 1) + (xi - xj + WIN - 1);
    dot += bias_table[rel * heads + head];
    if (shift > 0) {
      int r = wy * WIN + yj, c = wx * WIN + xj;
      int lr = (r < H - WIN) ? 0 : ((r < H - shift) ? 1 : 2);
      int lc = (c < Wd - WIN) ? 0 : ((c < Wd - shift) ? 1 : 2);
      if (lr * 3 + lc != li) dot -= 100.0f;
    }
    s[j] = dot;
    mx = fmaxf(mx, dot);
  }
  float se = 0.0f;
  for (int j = 0; j < 49; ++j) { s[j] = __expf(s[j] - mx); se += s[j]; }
  const float inv = 1.0f / se;

  const int ro = (wy * WIN + yi + shift) % H;
  const int co = (wx * WIN + xi + shift) % Wd;
  size_t obase = (size_t)((b * H + ro) * Wd + co) * C + head * 32;
  for (int d = 0; d < 32; ++d) {
    float o = 0.0f;
    for (int j = 0; j < 49; ++j) o += s[j] * vs[j * 32 + d];
    attn_out[obase + d] = o * inv;
  }
}

// ---------------------------------------------------------------------------
// Patch-embed unfold: x (16,3,224,224) -> xp (16*56*56, 48), f = c*16+dy*4+dx
// ---------------------------------------------------------------------------
__global__ void patch_gather_kernel(const float* __restrict__ x, float* __restrict__ xp)
{
  int idx = blockIdx.x * blockDim.x + threadIdx.x;
  if (idx >= 50176 * 48) return;
  int f   = idx % 48;
  int row = idx / 48;
  int pw  = row % 56;
  int ph  = (row / 56) % 56;
  int b   = row / (56 * 56);
  int c   = f / 16;
  int dy  = (f % 16) / 4;
  int dx  = f % 4;
  xp[idx] = x[((size_t)(b * 3 + c) * 224 + ph * 4 + dy) * 224 + pw * 4 + dx];
}

__global__ void meanpool_kernel(const float* __restrict__ in, float* __restrict__ out,
                                int T, int C)
{
  int b = blockIdx.x;
  for (int c = threadIdx.x; c < C; c += blockDim.x) {
    float s = 0.f;
    for (int t = 0; t < T; ++t) s += in[((size_t)b * T + t) * C + c];
    out[(size_t)b * C + c] = s / (float)T;
  }
}

// Small head layers (M=16): linear -> optional BN(eval) -> optional GELU.
__global__ void dense_kernel(const float* __restrict__ A, const float* __restrict__ W,
                             const float* __restrict__ bias, float* __restrict__ out,
                             int M, int N, int K,
                             const float* __restrict__ bng, const float* __restrict__ bnb,
                             const float* __restrict__ bnm, const float* __restrict__ bnv,
                             int do_gelu)
{
  int idx = blockIdx.x * blockDim.x + threadIdx.x;
  if (idx >= M * N) return;
  int m = idx / N, n = idx % N;
  float acc = bias[n];
  for (int k = 0; k < K; ++k) acc += A[(size_t)m * K + k] * W[(size_t)n * K + k];
  if (bng) acc = (acc - bnm[n]) * rsqrtf(bnv[n] + 1e-5f) * bng[n] + bnb[n];
  if (do_gelu) acc = gelu_exact(acc);
  out[idx] = acc;
}

// ---------------------------------------------------------------------------
// Orchestration
// ---------------------------------------------------------------------------
extern "C" void kernel_launch(void* const* d_in, const int* in_sizes, int n_in,
                              void* d_out, int out_size, void* d_ws, size_t ws_size,
                              hipStream_t stream)
{
  (void)in_sizes; (void)n_in; (void)out_size; (void)ws_size;
  int p = 0;
  auto nx = [&]() { return (const float*)d_in[p++]; };

  const float* x        = nx();
  const float* patch_w  = nx();
  const float* patch_b  = nx();
  const float* patch_g  = nx();
  const float* patch_be = nx();

  struct Blk {
    const float *ln1g, *ln1b, *qkvw, *qkvb, *projw, *projb, *bias,
                *ln2g, *ln2b, *fc1w, *fc1b, *fc2w, *fc2b;
  } blk[24];
  const int depths[4]    = {2, 2, 18, 2};
  const int heads_arr[4] = {4, 8, 16, 32};
  int bi = 0;
  for (int st = 0; st < 4; ++st)
    for (int d = 0; d < depths[st]; ++d) {
      Blk& bk = blk[bi++];
      bk.ln1g = nx(); bk.ln1b = nx();
      bk.qkvw = nx(); bk.qkvb = nx();
      bk.projw = nx(); bk.projb = nx();
      bk.bias = nx();
      bk.ln2g = nx(); bk.ln2b = nx();
      bk.fc1w = nx(); bk.fc1b = nx();
      bk.fc2w = nx(); bk.fc2b = nx();
    }
  struct Mg { const float *g, *b, *w; } mg[3];
  for (int i = 0; i < 3; ++i) { mg[i].g = nx(); mg[i].b = nx(); mg[i].w = nx(); }
  const float* norm_g = nx();
  const float* norm_b = nx();
  const float* w1  = nx(); const float* b1  = nx();
  const float* bn1g = nx(); const float* bn1b = nx();
  const float* bn1m = nx(); const float* bn1v = nx();
  const float* w2  = nx(); const float* b2  = nx();
  const float* bn2g = nx(); const float* bn2b = nx();
  const float* bn2m = nx(); const float* bn2v = nx();
  const float* w3  = nx(); const float* b3  = nx();

  // Workspace: h (50176x128) + A (50176x128) + B (50176x512) ~= 154 MB
  float* hbuf = (float*)d_ws;
  float* abuf = hbuf + (size_t)50176 * 128;
  float* bbuf = abuf + (size_t)50176 * 128;

  auto gemm = [&](const float* A, const float* W, const float* bias, const float* res,
                  float* out, int M, int N, int K, int flags) {
    dim3 grid((N + 127) / 128, (M + 63) / 64);
    wmma_gemm_kernel<<<grid, 256, 0, stream>>>(A, W, bias, res, out, M, N, K, flags);
  };
  auto lnorm = [&](const float* in, const float* g, const float* b, float* out,
                   int rows, int C) {
    layernorm_kernel<<<rows, 256, 0, stream>>>(in, g, b, out, C);
  };

  // --- patch embed: unfold -> 48x128 GEMM -> LN -> residual stream h ---
  patch_gather_kernel<<<(50176 * 48 + 255) / 256, 256, 0, stream>>>(x, bbuf);
  gemm(bbuf, patch_w, patch_b, nullptr, abuf, 50176, 128, 48, 0);
  lnorm(abuf, patch_g, patch_be, hbuf, 50176, 128);

  int Hc = 56, Wc = 56, Cc = 128;
  bi = 0;
  for (int st = 0; st < 4; ++st) {
    const int heads = heads_arr[st];
    const bool shiftable = Hc > WIN;
    const int M = 16 * Hc * Wc;
    for (int d = 0; d < depths[st]; ++d) {
      const Blk& bk = blk[bi++];
      const int shift = (d % 2 == 1 && shiftable) ? (WIN / 2) : 0;
      // attention branch
      lnorm(hbuf, bk.ln1g, bk.ln1b, abuf, M, Cc);
      gemm(abuf, bk.qkvw, bk.qkvb, nullptr, bbuf, M, 3 * Cc, Cc, 0);
      dim3 ag(16 * (Hc / WIN) * (Wc / WIN), heads);
      window_attn_kernel<<<ag, 64, 0, stream>>>(bbuf, bk.bias, abuf,
                                                Hc, Wc, Cc, heads, shift);
      gemm(abuf, bk.projw, bk.projb, hbuf, hbuf, M, Cc, Cc, 0);  // h += proj
      // MLP branch
      lnorm(hbuf, bk.ln2g, bk.ln2b, abuf, M, Cc);
      gemm(abuf, bk.fc1w, bk.fc1b, nullptr, bbuf, M, 4 * Cc, Cc, GEMM_GELU);
      gemm(bbuf, bk.fc2w, bk.fc2b, hbuf, hbuf, M, Cc, 4 * Cc, 0); // h += mlp
    }
    if (st < 3) {
      const int rows = 16 * (Hc / 2) * (Wc / 2);
      merge_ln_kernel<<<rows, 256, 0, stream>>>(hbuf, mg[st].g, mg[st].b, bbuf,
                                                Hc, Wc, Cc);
      gemm(bbuf, mg[st].w, nullptr, nullptr, hbuf, rows, 2 * Cc, 4 * Cc, 0);
      Hc /= 2; Wc /= 2; Cc *= 2;
    }
  }

  // --- final norm, global average pool, classifier head ---
  lnorm(hbuf, norm_g, norm_b, abuf, 16 * 49, 1024);
  float* pooled = bbuf;
  float* z1 = pooled + 16 * 1024;
  float* z2 = z1 + 16 * 512;
  meanpool_kernel<<<16, 256, 0, stream>>>(abuf, pooled, 49, 1024);
  dense_kernel<<<(16 * 512 + 255) / 256, 256, 0, stream>>>(
      pooled, w1, b1, z1, 16, 512, 1024, bn1g, bn1b, bn1m, bn1v, 1);
  dense_kernel<<<(16 * 256 + 255) / 256, 256, 0, stream>>>(
      z1, w2, b2, z2, 16, 256, 512, bn2g, bn2b, bn2m, bn2v, 1);
  dense_kernel<<<(16 * 100 + 255) / 256, 256, 0, stream>>>(
      z2, w3, b3, (float*)d_out, 16, 100, 256,
      nullptr, nullptr, nullptr, nullptr, 0);
}
